// EGNNExpert_20538533609914
// MI455X (gfx1250) — matrix operand order repaired and checked by reading, not compiled, not run on hardware
//
#include <hip/hip_runtime.h>
#include <hip/hip_bf16.h>
#include <math.h>

typedef __attribute__((ext_vector_type(16))) __bf16 v16bf;
typedef __attribute__((ext_vector_type(8)))  __bf16 v8bf;
typedef __attribute__((ext_vector_type(8)))  float  v8f;

#define N_NODES 50000
#define N_EDGES 800000
#define DIM_IN  64
#define HID     192
#define NH      12
#define HC      16
#define NLAYER  3
#define ETOT    (N_EDGES + N_NODES)
#define LN_EPS  1e-5f

// ---------------- helpers ----------------

__device__ __forceinline__ void atomicMaxF(float* addr, float val) {
  // IEEE float max via integer atomics (works for mixed signs, init = -inf)
  if (val >= 0.f) atomicMax((int*)addr, __float_as_int(val));
  else            atomicMin((unsigned int*)addr, (unsigned int)__float_as_int(val));
}

__device__ __forceinline__ int edge_src(int e, const int* __restrict__ src) {
  return (e < N_EDGES) ? src[e] : (e - N_EDGES);
}
__device__ __forceinline__ int edge_dst(int e, const int* __restrict__ dst) {
  return (e < N_EDGES) ? dst[e] : (e - N_EDGES);
}

// ---------------- misc small kernels ----------------

__global__ void cvt_bf16_kernel(const float* __restrict__ in, __bf16* __restrict__ out, int n) {
  int i = blockIdx.x * blockDim.x + threadIdx.x;
  if (i < n) out[i] = (__bf16)in[i];
}

__global__ void fill_kernel(float* __restrict__ p, float v, int n) {
  int i = blockIdx.x * blockDim.x + threadIdx.x;
  if (i < n) p[i] = v;
}

// accumulate per-dst count and edge_attr sum (for self-loop fill_value='mean')
__global__ void edge_mean_kernel(const int* __restrict__ dst, const float* __restrict__ eattr,
                                 float* __restrict__ cnt, float* __restrict__ msum) {
  int e = blockIdx.x * blockDim.x + threadIdx.x;
  if (e >= N_EDGES) return;
  int d = dst[e];
  atomicAdd(&cnt[d], 1.f);
  atomicAdd(&msum[2 * d + 0], eattr[2 * e + 0]);
  atomicAdd(&msum[2 * d + 1], eattr[2 * e + 1]);
}

__global__ void finalize_mean_kernel(const float* __restrict__ cnt, float* __restrict__ mattr) {
  int n = blockIdx.x * blockDim.x + threadIdx.x;
  if (n >= N_NODES) return;
  float c = fmaxf(cnt[n], 1.f);
  mattr[2 * n + 0] /= c;
  mattr[2 * n + 1] /= c;
}

// ---------------- WMMA weight pre-permutation ----------------
// B-fragment layout (16-bit B, KxN per 32-K tile): lane half h, col=lane&15;
// elems 0..7 <-> K=8h+i, elems 8..15 <-> K=16+8h+(i-8). Flat:
// Wp[(((t*kT + kt)*32 + lane)*16) + i] = W[kt*32 + (lane>>4)*8 + (i&7) + (i>>3)*16][t*16 + (lane&15)]
__global__ void convert_w_kernel(const float* __restrict__ W, __bf16* __restrict__ Wp, int K) {
  int idx = blockIdx.x * blockDim.x + threadIdx.x;
  if (idx >= K * HID) return;
  int kT   = K >> 5;
  int i    = idx & 15;
  int lane = (idx >> 4) & 31;
  int rest = idx >> 9;
  int kt   = rest % kT;
  int t    = rest / kT;
  int k    = kt * 32 + (lane >> 4) * 8 + (i & 7) + (i >> 3) * 16;
  int col  = t * 16 + (lane & 15);
  Wp[idx] = (__bf16)W[(size_t)k * HID + col];
}

// ---------------- GEMM: C[M x 192] = A_bf16[M x K] @ Wp + bias ----------------
// 12 waves per block; wave w owns output column tile w (cols 16w..16w+15);
// each block covers 16 rows. A-fragment per ISA 16-bit A layout:
// lane half h, row r: elems 0..7 = K(kt*32 + 8h + i), elems 8..15 = K(+16).
__global__ __launch_bounds__(384) void gemm_bf16_wmma(
    const __bf16* __restrict__ A, const __bf16* __restrict__ Wp,
    const float* __restrict__ bias, float* __restrict__ C, int M, int K) {
  const int lane = threadIdx.x & 31;
  const int wave = threadIdx.x >> 5;     // 0..11 == column tile
  const int half = lane >> 4;
  const int l16  = lane & 15;
  const int row0 = blockIdx.x << 4;
  const int kT   = K >> 5;

  int r = row0 + l16;
  if (r >= M) r = M - 1;                 // clamp keeps EXEC uniform for WMMA
  const __bf16* arow = A + (size_t)r * K + half * 8;
  const __bf16* wp   = Wp + (size_t)wave * kT * 512;   // 32*16 elems per K-tile

  v8f acc = {};
  for (int kt = 0; kt < kT; ++kt) {
    v8bf lo = *(const v8bf*)(arow + kt * 32);
    v8bf hi = *(const v8bf*)(arow + kt * 32 + 16);
    v16bf a = __builtin_shufflevector(lo, hi, 0,1,2,3,4,5,6,7,8,9,10,11,12,13,14,15);
    v16bf bm = *(const v16bf*)(wp + (size_t)kt * 512 + lane * 16);
    acc = __builtin_amdgcn_wmma_f32_16x16x32_bf16(false, a, false, bm,
                                                  (short)0, acc, false, false);
  }

  const int col = wave * 16 + l16;
  const float bb = bias[col];
#pragma unroll
  for (int v = 0; v < 8; ++v) {
    int rr = row0 + half * 8 + v;        // D layout: VGPR v -> M = 8*half + v
    if (rr < M) C[(size_t)rr * HID + col] = acc[v] + bb;
  }
}

// ---------------- row-wise epilogues with LayerNorm ----------------
// MODE 0: LN(gelu(xa)) ; MODE 1: LN(xa + xb + extra) ; MODE 2: LN(xa * sigmoid(xb))
template <int MODE>
__global__ __launch_bounds__(256) void rowln_kernel(
    const float* __restrict__ xa, const float* __restrict__ xb,
    const float* __restrict__ extra,
    const float* __restrict__ g, const float* __restrict__ b,
    float* __restrict__ out) {
  __shared__ float red[256];
  __shared__ float vals[HID];
  const int row = blockIdx.x, t = threadIdx.x;
  float v = 0.f;
  if (t < HID) {
    size_t i = (size_t)row * HID + t;
    if (MODE == 0) {
      float x = xa[i];
      v = 0.5f * x * (1.f + erff(x * 0.70710678118654752f));   // exact GELU
    } else if (MODE == 1) {
      v = xa[i] + xb[i] + extra[t];
    } else {
      float z = xb[i];
      v = xa[i] / (1.f + __expf(-z));
    }
    vals[t] = v;
  }
  red[t] = v; __syncthreads();
  for (int s = 128; s > 0; s >>= 1) { if (t < s) red[t] += red[t + s]; __syncthreads(); }
  const float mean = red[0] * (1.f / HID); __syncthreads();
  float d = (t < HID) ? vals[t] - mean : 0.f;
  red[t] = d * d; __syncthreads();
  for (int s = 128; s > 0; s >>= 1) { if (t < s) red[t] += red[t + s]; __syncthreads(); }
  const float inv = rsqrtf(red[0] * (1.f / HID) + LN_EPS);
  if (t < HID) out[(size_t)row * HID + t] = d * inv * g[t] + b[t];
}

// ---------------- GATv2 edge passes (one thread per (edge, head)) ----------------

__global__ void edge_score_kernel(const int* __restrict__ src, const int* __restrict__ dst,
    const float* __restrict__ eattr, const float* __restrict__ mattr,
    const float* __restrict__ xl, const float* __restrict__ xr,
    const float* __restrict__ We, const float* __restrict__ att,
    float* __restrict__ s, float* __restrict__ smax) {
  int idx = blockIdx.x * blockDim.x + threadIdx.x;
  if (idx >= ETOT * NH) return;
  int h = idx % NH, e = idx / NH;
  int si, di; float a0, a1;
  if (e < N_EDGES) { si = src[e]; di = dst[e]; a0 = eattr[2*e]; a1 = eattr[2*e+1]; }
  else             { si = di = e - N_EDGES;    a0 = mattr[2*si]; a1 = mattr[2*si+1]; }
  const float* xls = xl + (size_t)si * HID + h * HC;
  const float* xrd = xr + (size_t)di * HID + h * HC;
  const float* w0  = We + h * HC;
  const float* w1  = We + HID + h * HC;
  const float* at  = att + h * HC;
  float acc = 0.f;
#pragma unroll
  for (int c = 0; c < HC; ++c) {
    float m = xls[c] + xrd[c] + a0 * w0[c] + a1 * w1[c];
    m = (m > 0.f) ? m : 0.2f * m;                 // leaky_relu 0.2
    acc += m * at[c];
  }
  s[idx] = acc;
  atomicMaxF(&smax[(size_t)di * NH + h], acc);
}

__global__ void edge_exp_kernel(const int* __restrict__ dst,
    float* __restrict__ s, const float* __restrict__ smax, float* __restrict__ den) {
  int idx = blockIdx.x * blockDim.x + threadIdx.x;
  if (idx >= ETOT * NH) return;
  int h = idx % NH, e = idx / NH;
  int di = edge_dst(e, dst);
  float ex = __expf(s[idx] - smax[(size_t)di * NH + h]);
  s[idx] = ex;
  atomicAdd(&den[(size_t)di * NH + h], ex);
}

__global__ void edge_aggr_kernel(const int* __restrict__ src, const int* __restrict__ dst,
    const float* __restrict__ s, const float* __restrict__ den,
    const float* __restrict__ xl, float* __restrict__ conv) {
  int idx = blockIdx.x * blockDim.x + threadIdx.x;
  if (idx >= ETOT * NH) return;
  int h = idx % NH, e = idx / NH;
  int si = edge_src(e, src), di = edge_dst(e, dst);
  float alpha = s[idx] / den[(size_t)di * NH + h];
  const float* xls = xl + (size_t)si * HID + h * HC;
  float* o = conv + (size_t)di * HID + h * HC;
#pragma unroll
  for (int c = 0; c < HC; ++c) atomicAdd(&o[c], alpha * xls[c]);
}

// ---------------- host orchestration ----------------

extern "C" void kernel_launch(void* const* d_in, const int* in_sizes, int n_in,
                              void* d_out, int out_size, void* d_ws, size_t ws_size,
                              hipStream_t stream) {
  const float* x      = (const float*)d_in[0];
  const int*   ei     = (const int*)  d_in[1];
  const float* eattr  = (const float*)d_in[2];
  const float* W_in   = (const float*)d_in[3];
  const float* b_in   = (const float*)d_in[4];
  const float* g_lni  = (const float*)d_in[5];
  const float* b_lni  = (const float*)d_in[6];
  const float* Wl     = (const float*)d_in[7];
  const float* bl     = (const float*)d_in[8];
  const float* Wr     = (const float*)d_in[9];
  const float* br     = (const float*)d_in[10];
  const float* We     = (const float*)d_in[11];
  const float* att    = (const float*)d_in[12];
  const float* bo     = (const float*)d_in[13];
  const float* g_res  = (const float*)d_in[14];
  const float* b_res  = (const float*)d_in[15];
  const float* Wg     = (const float*)d_in[16];
  const float* bg     = (const float*)d_in[17];
  const float* g_f    = (const float*)d_in[18];
  const float* b_f    = (const float*)d_in[19];
  const int* src = ei;
  const int* dst = ei + N_EDGES;

  char* base = (char*)d_ws;
  size_t off = 0;
  auto alloc = [&](size_t bytes) -> void* {
    void* p = base + off;
    off += bytes; off = (off + 255) & ~(size_t)255;
    return p;
  };
  float*  h     = (float*) alloc((size_t)N_NODES * HID * 4);
  float*  tmp   = (float*) alloc((size_t)N_NODES * HID * 4);
  float*  xl    = (float*) alloc((size_t)N_NODES * HID * 4);
  float*  xr    = (float*) alloc((size_t)N_NODES * HID * 4);
  __bf16* hbf   = (__bf16*)alloc((size_t)N_NODES * HID * 2);
  float*  s     = (float*) alloc((size_t)ETOT * NH * 4);
  float*  smax  = (float*) alloc((size_t)N_NODES * NH * 4);
  float*  den   = (float*) alloc((size_t)N_NODES * NH * 4);
  float*  cnt   = (float*) alloc((size_t)N_NODES * 4);
  float*  mattr = (float*) alloc((size_t)N_NODES * 2 * 4);
  __bf16* WpA   = (__bf16*)alloc((size_t)HID * HID * 2);
  __bf16* WpB   = (__bf16*)alloc((size_t)HID * HID * 2);

  const int TB = 256;
  const int gEdge  = (N_EDGES + TB - 1) / TB;
  const int gEH    = (ETOT * NH + TB - 1) / TB;
  const int gNode  = (N_NODES + TB - 1) / TB;
  const int gHidAll= ((int)((size_t)N_NODES * HID) + TB - 1) / TB;
  const int gemmGrid = N_NODES / 16;     // 3125, exact

  // self-loop mean edge_attr
  hipMemsetAsync(cnt,   0, (size_t)N_NODES * 4, stream);
  hipMemsetAsync(mattr, 0, (size_t)N_NODES * 8, stream);
  edge_mean_kernel<<<gEdge, TB, 0, stream>>>(dst, eattr, cnt, mattr);
  finalize_mean_kernel<<<gNode, TB, 0, stream>>>(cnt, mattr);

  // input projection: h = LN(gelu(x @ W_in + b_in))
  cvt_bf16_kernel<<<(N_NODES * DIM_IN + TB - 1) / TB, TB, 0, stream>>>(x, hbf, N_NODES * DIM_IN);
  convert_w_kernel<<<(DIM_IN * HID + TB - 1) / TB, TB, 0, stream>>>(W_in, WpA, DIM_IN);
  gemm_bf16_wmma<<<gemmGrid, 384, 0, stream>>>(hbf, WpA, b_in, tmp, N_NODES, DIM_IN);
  rowln_kernel<0><<<N_NODES, 256, 0, stream>>>(tmp, nullptr, nullptr, g_lni, b_lni, h);

  for (int l = 0; l < NLAYER; ++l) {
    cvt_bf16_kernel<<<gHidAll, TB, 0, stream>>>(h, hbf, N_NODES * HID);
    convert_w_kernel<<<(HID * HID + TB - 1) / TB, TB, 0, stream>>>(Wl + (size_t)l * HID * HID, WpA, HID);
    convert_w_kernel<<<(HID * HID + TB - 1) / TB, TB, 0, stream>>>(Wr + (size_t)l * HID * HID, WpB, HID);
    gemm_bf16_wmma<<<gemmGrid, 384, 0, stream>>>(hbf, WpA, bl + (size_t)l * HID, xl, N_NODES, HID);
    gemm_bf16_wmma<<<gemmGrid, 384, 0, stream>>>(hbf, WpB, br + (size_t)l * HID, xr, N_NODES, HID);

    fill_kernel<<<(N_NODES * NH + TB - 1) / TB, TB, 0, stream>>>(smax, -INFINITY, N_NODES * NH);
    hipMemsetAsync(den, 0, (size_t)N_NODES * NH * 4, stream);
    hipMemsetAsync(tmp, 0, (size_t)N_NODES * HID * 4, stream);

    edge_score_kernel<<<gEH, TB, 0, stream>>>(src, dst, eattr, mattr, xl, xr,
                                              We + (size_t)l * 2 * HID,
                                              att + (size_t)l * NH * HC, s, smax);
    edge_exp_kernel<<<gEH, TB, 0, stream>>>(dst, s, smax, den);
    edge_aggr_kernel<<<gEH, TB, 0, stream>>>(src, dst, s, den, xl, tmp);

    rowln_kernel<1><<<N_NODES, 256, 0, stream>>>(h, tmp, bo + (size_t)l * HID,
                                                 g_res + (size_t)l * HID,
                                                 b_res + (size_t)l * HID, h);
  }

  // shock gate + final LN
  cvt_bf16_kernel<<<gHidAll, TB, 0, stream>>>(h, hbf, N_NODES * HID);
  convert_w_kernel<<<(HID * HID + TB - 1) / TB, TB, 0, stream>>>(Wg, WpA, HID);
  gemm_bf16_wmma<<<gemmGrid, 384, 0, stream>>>(hbf, WpA, bg, tmp, N_NODES, HID);
  rowln_kernel<2><<<N_NODES, 256, 0, stream>>>(h, tmp, nullptr, g_f, b_f, (float*)d_out);
}
